// LSTM_36825049596199
// MI455X (gfx1250) — matrix operand order repaired and checked by reading, not compiled, not run on hardware
//
#include <hip/hip_runtime.h>
#include <hip/hip_bf16.h>
#include <cstdint>

// ---------------------------------------------------------------------------
// Bidirectional 3-layer LSTM for MI455X (gfx1250, wave32, WMMA).
//   B=64, T=512, H=128, EMB=128, gates 4H=512.
// Phase 1: embedding gather -> f16 activations.
// Phase 2: per layer: big WMMA GEMM for x @ w_ih^T (+biases) -> fp32 xg.
// Phase 3: per layer: persistent-workgroup recurrent scan (1 WGP per
//          direction) with w_hh + h + gate staging resident in 272KB LDS,
//          v_wmma_f32_16x16x32_f16 for h @ w_hh^T each timestep.
// Phase 4: tiny linear head on last timestep.
// ---------------------------------------------------------------------------

typedef __attribute__((ext_vector_type(16))) _Float16 v16h;
typedef __attribute__((ext_vector_type(8)))  _Float16 v8h;
typedef __attribute__((ext_vector_type(8)))  float    v8f;

#define EMBD  128
#define Hdim  128
#define Bsz   64
#define Tlen  512
#define G4    512                 // 4*H (gate width)
#define ROWS  (Bsz * Tlen)        // 32768

// 272KB dynamic LDS for the scan kernel (CDNA5: 320KB/WGP)
#define W_LDS_BYTES   (G4 * Hdim * 2)          // 131072: w_hh as f16 [512][128]
#define H_LDS_BYTES   (Bsz * Hdim * 2)         //  16384: h as f16 [64][128]
#define G_LDS_BYTES   (Bsz * G4 * 4)           // 131072: pre-activation gates fp32
#define SCAN_LDS      (W_LDS_BYTES + H_LDS_BYTES + G_LDS_BYTES)

__device__ __forceinline__ v8f wmma_f16f32(v16h a, v16h b, v8f c) {
  // D = A(16x32 f16) * B(32x16 f16) + C(16x16 f32)
  return __builtin_amdgcn_wmma_f32_16x16x32_f16(
      /*neg_a=*/false, a, /*neg_b=*/false, b,
      /*c_mod=*/(short)0, c, /*reuse_a=*/false, /*reuse_b=*/false);
}

// A fragment: 16x32 f16 tile at (m0, k0), row-major source with leading dim lda.
// ISA layout: lane m = m0 + (lane&15); half = lane>>4;
//   elems 0..7  = K[k0 + half*8 .. +7],  elems 8..15 = K[k0+16+half*8 .. +7]
__device__ __forceinline__ v16h load_a_frag(const _Float16* __restrict__ A,
                                            int lda, int m0, int k0, int lane) {
  const int m = m0 + (lane & 15);
  const int half = lane >> 4;
  const _Float16* p = A + (size_t)m * lda + k0 + half * 8;
  v8h lo = *(const v8h*)(p);
  v8h hi = *(const v8h*)(p + 16);
  v16h r;
#pragma unroll
  for (int i = 0; i < 8; ++i) { r[i] = lo[i]; r[8 + i] = hi[i]; }
  return r;
}

// B fragment: 32x16 f16 tile, where B[k][n] = W[n][k] (W row-major, ld = ldw).
// ISA layout: lane n = n0 + (lane&15); half = lane>>4;
//   elems 0..7  = K[k0 + half*8 .. +7],  elems 8..15 = K[k0+16+half*8 .. +7]
__device__ __forceinline__ v16h load_b_frag(const _Float16* __restrict__ W,
                                            int ldw, int n0, int k0, int lane) {
  const int n = n0 + (lane & 15);
  const int half = lane >> 4;
  const _Float16* p = W + (size_t)n * ldw + k0 + half * 8;
  v8h lo = *(const v8h*)(p);
  v8h hi = *(const v8h*)(p + 16);
  v16h r;
#pragma unroll
  for (int i = 0; i < 8; ++i) { r[i] = lo[i]; r[8 + i] = hi[i]; }
  return r;
}

__device__ __forceinline__ float sigf(float x) {
  return 1.0f / (1.0f + __expf(-x));
}

// ---------------------------------------------------------------------------
// Embedding gather: x0[b*T+t][k] = (f16) emb[tok][k]
__global__ void __launch_bounds__(128)
lstm_embed_kernel(const int* __restrict__ X, const float* __restrict__ emb,
                  _Float16* __restrict__ x0) {
  const int row = blockIdx.x;           // b*T + t
  const int k = threadIdx.x;            // 0..127
  const int tok = X[row];
  x0[(size_t)row * EMBD + k] = (_Float16)emb[(size_t)tok * EMBD + k];
}

// fp32 -> f16 weight conversion
__global__ void __launch_bounds__(256)
lstm_cvt_kernel(const float* __restrict__ in, _Float16* __restrict__ out, int n) {
  int i = blockIdx.x * 256 + threadIdx.x;
  if (i < n) out[i] = (_Float16)in[i];
}

// ---------------------------------------------------------------------------
// Input projection: XG[dir][m][n] = sum_k X[m][k]*Wih[dir][n][k] + bih + bhh
// grid (ROWS/16, 8, 2), block 128 (4 waves; wave w handles N-tile bIdx.y*4+w)
__global__ void __launch_bounds__(128)
lstm_gemm_kernel(const _Float16* __restrict__ X, int Din,
                 const _Float16* __restrict__ Wih,       // [2][512][Din] f16
                 const float* __restrict__ bih,          // [2][512]
                 const float* __restrict__ bhh,          // [2][512]
                 float* __restrict__ XG) {               // [2][ROWS][512]
  const int dir  = blockIdx.z;
  const int mt   = blockIdx.x;                  // M tile: rows mt*16..+15
  const int wave = threadIdx.x >> 5;
  const int lane = threadIdx.x & 31;
  const int nt   = blockIdx.y * 4 + wave;       // N tile 0..31

  const _Float16* W = Wih + (size_t)dir * G4 * Din;
  v8f acc = {};
  const int kchunks = Din >> 5;
#pragma unroll 4
  for (int kc = 0; kc < kchunks; ++kc) {
    v16h a = load_a_frag(X, Din, mt * 16, kc * 32, lane);
    v16h b = load_b_frag(W, Din, nt * 16, kc * 32, lane);
    acc = wmma_f16f32(a, b, acc);
  }

  const int n = nt * 16 + (lane & 15);
  const float bias = bih[dir * G4 + n] + bhh[dir * G4 + n];
  float* out = XG + (size_t)dir * ROWS * G4;
  const int half = lane >> 4;
#pragma unroll
  for (int r = 0; r < 8; ++r) {
    const int m = mt * 16 + r + 8 * half;       // C layout: M = r + 8*(lane/16)
    out[(size_t)m * G4 + n] = acc[r] + bias;
  }
}

// ---------------------------------------------------------------------------
// Recurrent scan: one persistent workgroup (32 waves) per direction.
// grid(2), block(1024), dynamic LDS = SCAN_LDS.
__global__ void __launch_bounds__(1024)
lstm_scan_kernel(const float* __restrict__ xg,        // [2][ROWS][512] (this layer)
                 const _Float16* __restrict__ whh16,  // [2][512][128] f16
                 const float* __restrict__ hx_l,      // [2][64][128] (layer slice)
                 const float* __restrict__ cx_l,      // [2][64][128]
                 _Float16* __restrict__ y) {          // [ROWS][256] = [B,T,2H]
  extern __shared__ char smem[];
  _Float16* w_lds = (_Float16*)smem;                               // [512][128]
  _Float16* h_lds = (_Float16*)(smem + W_LDS_BYTES);               // [64][128]
  float*    g_lds = (float*)(smem + W_LDS_BYTES + H_LDS_BYTES);    // [64][512]

  const int dir  = blockIdx.x;       // 0 = fwd, 1 = bwd (reverse time)
  const int tid  = threadIdx.x;
  const int wave = tid >> 5;         // 0..31  -> owns N tile nt = wave
  const int lane = tid & 31;

  // --- stage w_hh into LDS (128KB of f16 = 32K dwords, 32 per thread)
  {
    const uint32_t* src = (const uint32_t*)(whh16 + (size_t)dir * G4 * Hdim);
    uint32_t* dst = (uint32_t*)w_lds;
#pragma unroll
    for (int i = 0; i < (G4 * Hdim / 2) / 1024; ++i)
      dst[tid + i * 1024] = src[tid + i * 1024];
  }

  // --- per-thread ownership of 8 (b,u) cells for the elementwise phase
  const int eb = tid >> 4;               // 0..63
  const int eu = (tid & 15) * 8;         // 0..120
  float creg[8];
  {
    const float* h0 = hx_l + (size_t)dir * Bsz * Hdim;
    const float* c0 = cx_l + (size_t)dir * Bsz * Hdim;
#pragma unroll
    for (int j = 0; j < 8; ++j) {
      creg[j] = c0[eb * Hdim + eu + j];
      h_lds[eb * Hdim + eu + j] = (_Float16)h0[eb * Hdim + eu + j];
    }
  }
  __syncthreads();

  const float* xgd = xg + (size_t)dir * ROWS * G4;
  const int nt = wave;                   // N tile owned by this wave
  const int ncol = nt * 16 + (lane & 15);
  const int half = lane >> 4;

  for (int t = 0; t < Tlen; ++t) {
    const int tcur = dir ? (Tlen - 1 - t) : t;

    // ---- g = h @ w_hh^T  (4 M-tiles x 1 N-tile per wave, K = 128)
    v8f acc[4];
#pragma unroll
    for (int mt = 0; mt < 4; ++mt) acc[mt] = (v8f){};
#pragma unroll
    for (int kc = 0; kc < 4; ++kc) {
      v16h bf = load_b_frag(w_lds, Hdim, nt * 16, kc * 32, lane);
#pragma unroll
      for (int mt = 0; mt < 4; ++mt) {
        v16h af = load_a_frag(h_lds, Hdim, mt * 16, kc * 32, lane);
        acc[mt] = wmma_f16f32(af, bf, acc[mt]);
      }
    }
    // ---- add precomputed input projection, stage pre-activations in LDS
#pragma unroll
    for (int mt = 0; mt < 4; ++mt) {
#pragma unroll
      for (int r = 0; r < 8; ++r) {
        const int b = mt * 16 + r + 8 * half;
        const float v = acc[mt][r] + xgd[((size_t)b * Tlen + tcur) * G4 + ncol];
        g_lds[b * G4 + ncol] = v;
      }
    }
    __syncthreads();

    // ---- gates (torch order i,f,g,o), update c (regs) and h (LDS f16)
    {
      const float* gb = g_lds + eb * G4;
#pragma unroll
      for (int j = 0; j < 8; ++j) {
        const int u = eu + j;
        const float ig = gb[u];
        const float fg = gb[Hdim + u];
        const float gg = gb[2 * Hdim + u];
        const float og = gb[3 * Hdim + u];
        const float cn = sigf(fg) * creg[j] + sigf(ig) * tanhf(gg);
        creg[j] = cn;
        const float hn = sigf(og) * tanhf(cn);
        h_lds[eb * Hdim + u] = (_Float16)hn;
        y[((size_t)eb * Tlen + tcur) * (2 * Hdim) + dir * Hdim + u] = (_Float16)hn;
      }
    }
    // prefetch next timestep's xg rows (global_prefetch_b8)
    const int tn = dir ? (tcur - 1) : (tcur + 1);
    if (tn >= 0 && tn < Tlen) {
      const int pb = tid >> 4, pc = tid & 15;
      __builtin_prefetch(xgd + ((size_t)pb * Tlen + tn) * G4 + pc * 32, 0, 0);
    }
    __syncthreads();
  }
}

// ---------------------------------------------------------------------------
// Linear head on last timestep: out[b] = y[b, T-1, :] . lin_w + lin_b
__global__ void __launch_bounds__(64)
lstm_head_kernel(const _Float16* __restrict__ y, const float* __restrict__ lw,
                 const float* __restrict__ lb, float* __restrict__ out) {
  const int b = threadIdx.x;
  const _Float16* p = y + ((size_t)b * Tlen + (Tlen - 1)) * (2 * Hdim);
  float s = lb[0];
#pragma unroll 8
  for (int j = 0; j < 2 * Hdim; ++j) s += (float)p[j] * lw[j];
  out[b] = s;
}

// ---------------------------------------------------------------------------
extern "C" void kernel_launch(void* const* d_in, const int* in_sizes, int n_in,
                              void* d_out, int out_size, void* d_ws, size_t ws_size,
                              hipStream_t stream) {
  const int*   X     = (const int*)d_in[0];
  const float* emb   = (const float*)d_in[1];
  const float* hx    = (const float*)d_in[2];
  const float* cx    = (const float*)d_in[3];
  const float* lin_w = (const float*)d_in[4];
  const float* lin_b = (const float*)d_in[5];
  const float* w_ih[3] = {(const float*)d_in[6],  (const float*)d_in[10], (const float*)d_in[14]};
  const float* w_hh[3] = {(const float*)d_in[7],  (const float*)d_in[11], (const float*)d_in[15]};
  const float* b_ih[3] = {(const float*)d_in[8],  (const float*)d_in[12], (const float*)d_in[16]};
  const float* b_hh[3] = {(const float*)d_in[9],  (const float*)d_in[13], (const float*)d_in[17]};

  // ---- workspace layout
  char* ws = (char*)d_ws;
  size_t off = 0;
  float*    xg = (float*)(ws + off);    off += (size_t)2 * ROWS * G4 * 4;   // 128MB
  _Float16* x0 = (_Float16*)(ws + off); off += (size_t)ROWS * EMBD * 2;     // 8MB
  _Float16* xa = (_Float16*)(ws + off); off += (size_t)ROWS * 2 * Hdim * 2; // 16MB
  _Float16* xb = (_Float16*)(ws + off); off += (size_t)ROWS * 2 * Hdim * 2; // 16MB
  const int wih_elems[3] = {2 * G4 * EMBD, 2 * G4 * 2 * Hdim, 2 * G4 * 2 * Hdim};
  _Float16* wih16[3];
  _Float16* whh16[3];
  for (int l = 0; l < 3; ++l) { wih16[l] = (_Float16*)(ws + off); off += (size_t)wih_elems[l] * 2; }
  for (int l = 0; l < 3; ++l) { whh16[l] = (_Float16*)(ws + off); off += (size_t)2 * G4 * Hdim * 2; }

  // allow >64KB dynamic LDS for the scan kernel (272KB of CDNA5's 320KB/WGP)
  (void)hipFuncSetAttribute((const void*)lstm_scan_kernel,
                            hipFuncAttributeMaxDynamicSharedMemorySize, SCAN_LDS);

  // ---- phase 1: embedding + weight conversion
  lstm_embed_kernel<<<ROWS, 128, 0, stream>>>(X, emb, x0);
  for (int l = 0; l < 3; ++l) {
    lstm_cvt_kernel<<<(wih_elems[l] + 255) / 256, 256, 0, stream>>>(w_ih[l], wih16[l], wih_elems[l]);
    lstm_cvt_kernel<<<(2 * G4 * Hdim + 255) / 256, 256, 0, stream>>>(w_hh[l], whh16[l], 2 * G4 * Hdim);
  }

  const dim3 ggrid(ROWS / 16, 8, 2);
  const int Din[3] = {EMBD, 2 * Hdim, 2 * Hdim};
  const _Float16* xin[3] = {x0, xa, xb};
  _Float16* xout[3] = {xa, xb, xa};

  // ---- phases 2+3: per layer (GEMM fills the chip; scan = 2 persistent WGPs)
  for (int l = 0; l < 3; ++l) {
    lstm_gemm_kernel<<<ggrid, 128, 0, stream>>>(xin[l], Din[l], wih16[l],
                                                b_ih[l], b_hh[l], xg);
    lstm_scan_kernel<<<2, 1024, SCAN_LDS, stream>>>(xg, whh16[l],
                                                    hx + (size_t)2 * l * Bsz * Hdim,
                                                    cx + (size_t)2 * l * Bsz * Hdim,
                                                    xout[l]);
  }

  // ---- phase 4: head
  lstm_head_kernel<<<1, 64, 0, stream>>>(xa, lin_w, lin_b, (float*)d_out);
}
